// LocalWindowAttention_28913719837101
// MI455X (gfx1250) — matrix operand order repaired and checked by reading, not compiled, not run on hardware
//
#include <hip/hip_runtime.h>
#include <hip/hip_bf16.h>

typedef __attribute__((ext_vector_type(16))) _Float16 v16h;
typedef __attribute__((ext_vector_type(8)))  float    v8f;

union Frag { v16h h; unsigned u[8]; uint4 q[2]; };

static __device__ __forceinline__ unsigned pk2h(float a, float b) {
  union { _Float16 h[2]; unsigned u; } z;
  z.h[0] = (_Float16)a; z.h[1] = (_Float16)b;
  return z.u;
}

static __device__ __forceinline__ void loadB(Frag& f, const uint4* base,
                                             int step, int lane) {
  const uint4* p = base + ((size_t)step * 32 + lane) * 2;
  f.q[0] = p[0];
  f.q[1] = p[1];
}

// Opaque zero, provably a multiple of 32 elements (64 B): defeats LICM (the
// value is loop-variant as far as the compiler knows) without destroying
// alignment knowledge, so LDS reads still merge into ds_load_b128.
static __device__ __forceinline__ int opaque_zero_aligned() {
  int z = 0;
  asm volatile("" : "+v"(z));
  return z & ~31;
}

#define WMMA_F16(A, B, C) \
  __builtin_amdgcn_wmma_f32_16x16x32_f16(false, (A), false, (B), (short)0, (C), false, false)

// ---------------------------------------------------------------------------
// Weight pre-swizzle: emit f16 B-fragments in this kernel's per-lane order (contiguous k, same as the A loads).
//   frag(nt, kt, lane L, dword v) = pk( W[e*256+d], W[e*256+d+1] )
//   e = nt*16 + (L&15),  d = kt*32 + (L>>4)*16 + 2v
// ---------------------------------------------------------------------------
__global__ void pack_w_kernel(const float* __restrict__ W, unsigned* __restrict__ dst,
                              int ntiles) {
  const int t  = blockIdx.x * 32 + threadIdx.x;   // frag-lane id
  const int L  = t & 31;
  const int kt = (t >> 5) & 7;
  const int nt = t >> 8;
  if (nt >= ntiles) return;
  const int e = nt * 16 + (L & 15);
  const int dbase = kt * 32 + (L >> 4) * 16;
  const float* wrow = W + (size_t)e * 256 + dbase;
  unsigned u[8];
#pragma unroll
  for (int v = 0; v < 8; ++v) u[v] = pk2h(wrow[2 * v], wrow[2 * v + 1]);
  uint4* o = (uint4*)(dst + (size_t)t * 8);
  o[0] = make_uint4(u[0], u[1], u[2], u[3]);
  o[1] = make_uint4(u[4], u[5], u[6], u[7]);
}

// Expand relative-position bias to dense (8, 64, 64) f32.
__global__ void pack_bias_kernel(const float* __restrict__ tab,
                                 const int* __restrict__ idx,
                                 float* __restrict__ dst) {
  const int t  = blockIdx.x * 256 + threadIdx.x;  // 0..32767
  const int h  = t >> 12;
  const int pq = t & 4095;
  dst[t] = tab[idx[pq] * 8 + h];
}

// ---------------------------------------------------------------------------
// One workgroup per window (64 tokens x 256 dims). 256 threads = 8 waves.
// 144 KB LDS + <=256 VGPRs -> two workgroups per WGP for latency hiding.
// ---------------------------------------------------------------------------
__global__ __launch_bounds__(256, 4) void lwa_kernel(
    const float* __restrict__ x,
    const float* __restrict__ noise,
    const float* __restrict__ gamma,
    const float* __restrict__ beta,
    const uint4* __restrict__ wqkv,   // pre-swizzled qkv_w frags (48 x 8 x 32 x 2)
    const uint4* __restrict__ wproj,  // pre-swizzled proj_w frags (16 x 8 x 32 x 2)
    const float* __restrict__ proj_b,
    const float* __restrict__ biasd,  // dense (8,64,64)
    float* __restrict__ out)
{
  __shared__ __attribute__((aligned(16))) _Float16 sX [64 * 256];
  __shared__ __attribute__((aligned(16))) _Float16 sQ [64 * 256];
  __shared__ __attribute__((aligned(16))) _Float16 sK [64 * 256];
  __shared__ __attribute__((aligned(16))) _Float16 sVT[256 * 64];
  __shared__ __attribute__((aligned(16))) _Float16 sAS[8 * 16 * 64];

  const int tid  = threadIdx.x;
  const int lane = tid & 31;
  const int wvs  = __builtin_amdgcn_readfirstlane(tid >> 5); // wave id (SGPR)
  const int l16  = lane & 15;
  const int lh   = lane >> 4;
  const int w    = blockIdx.x;
  const int bb_  = w >> 8;

  const float* xg = x     + (size_t)w   * 16384;
  const float* ng = noise + (size_t)bb_ * 16384;

  // ---------------- Stage 1: xw = x*noise, LayerNorm(256) -> f16 sX ---------
  {
    const int p  = tid >> 2;
    const int d0 = (tid & 3) * 64;
    float xv[64];
    float s = 0.f, s2 = 0.f;
#pragma unroll
    for (int i = 0; i < 16; ++i) {
      float4 a = *(const float4*)(xg + p * 256 + d0 + i * 4);
      float4 z = *(const float4*)(ng + p * 256 + d0 + i * 4);
      float v0 = a.x * z.x, v1 = a.y * z.y, v2 = a.z * z.z, v3 = a.w * z.w;
      xv[i * 4 + 0] = v0; xv[i * 4 + 1] = v1;
      xv[i * 4 + 2] = v2; xv[i * 4 + 3] = v3;
      s  += v0 + v1 + v2 + v3;
      s2 += v0 * v0 + v1 * v1 + v2 * v2 + v3 * v3;
    }
    s  += __shfl_xor(s, 1);  s  += __shfl_xor(s, 2);
    s2 += __shfl_xor(s2, 1); s2 += __shfl_xor(s2, 2);
    const float mu  = s * (1.f / 256.f);
    const float var = s2 * (1.f / 256.f) - mu * mu;
    const float ri  = rsqrtf(var + 1e-5f);
#pragma unroll
    for (int i = 0; i < 32; ++i) {
      const int d = d0 + 2 * i;
      float y0 = (xv[2 * i + 0] - mu) * ri * gamma[d + 0] + beta[d + 0];
      float y1 = (xv[2 * i + 1] - mu) * ri * gamma[d + 1] + beta[d + 1];
      *(unsigned*)&sX[p * 256 + d] = pk2h(y0, y1);
    }
  }
  __syncthreads();

  // ---------------- Stage 2: qkv = xn @ qkv_w^T (f16 WMMA) ------------------
  // Wave-contiguous B-fragment stream: frag index = wvs*48 + nti*8 + kt.
  // Depth-4 software pipeline on the weight fragments.
  const float kscale = 0.17677669529663687f; // 1/sqrt(32), folded into q
  {
    Frag bbuf[4];
#pragma unroll
    for (int i = 0; i < 4; ++i) loadB(bbuf[i], wqkv, wvs * 48 + i, lane);
    for (int nti = 0; nti < 6; ++nti) {
      const int nt = wvs * 6 + nti;         // SGPR-uniform
      const int e  = nt * 16 + l16;
      const _Float16* sxl = sX + opaque_zero_aligned();  // block A-frag hoisting
      v8f acc[4];
#pragma unroll
      for (int m = 0; m < 4; ++m) acc[m] = {};
#pragma unroll
      for (int kt = 0; kt < 8; ++kt) {
#pragma unroll
        for (int m = 0; m < 4; ++m) {
          Frag af;
          const _Float16* arow = sxl + (m * 16 + l16) * 256 + kt * 32;
#pragma unroll
          for (int v = 0; v < 8; ++v) {
            const int kk = lh * 16 + 2 * v;
            af.u[v] = *(const unsigned*)(arow + kk);
          }
          acc[m] = WMMA_F16(af.h, bbuf[kt & 3].h, acc[m]);
        }
        int s = nti * 8 + kt + 4;           // prefetch distance 4
        if (s > 47) s = 47;
        loadB(bbuf[kt & 3], wqkv, wvs * 48 + s, lane);
      }
      // Scatter q/k/v (scalar branch: nt is an SGPR value).
      if (nt < 16) {
#pragma unroll
        for (int m = 0; m < 4; ++m)
#pragma unroll
          for (int j = 0; j < 8; ++j)
            sQ[(m * 16 + j + lh * 8) * 256 + e] = (_Float16)(acc[m][j] * kscale);
      } else if (nt < 32) {
#pragma unroll
        for (int m = 0; m < 4; ++m)
#pragma unroll
          for (int j = 0; j < 8; ++j)
            sK[(m * 16 + j + lh * 8) * 256 + (e - 256)] = (_Float16)acc[m][j];
      } else {
        // tokens are contiguous in sVT: pack pairs along j into b32 stores
#pragma unroll
        for (int m = 0; m < 4; ++m)
#pragma unroll
          for (int j = 0; j < 8; j += 2)
            *(unsigned*)&sVT[(e - 512) * 64 + m * 16 + lh * 8 + j] =
                pk2h(acc[m][j], acc[m][j + 1]);
      }
    }
  }
  __syncthreads();

  // ---------------- Stage 3: attention, one head per wave -------------------
  {
    const int h = wvs;
    v8f sc[4][4];
    Frag qf[4];
#pragma unroll
    for (int tm = 0; tm < 4; ++tm) {
      const _Float16* qrow = sQ + (tm * 16 + l16) * 256 + h * 32;
#pragma unroll
      for (int v = 0; v < 8; ++v) {
        const int kk = lh * 16 + 2 * v;
        qf[tm].u[v] = *(const unsigned*)(qrow + kk);
      }
    }
#pragma unroll
    for (int tn = 0; tn < 4; ++tn) {
      Frag kf;
      const _Float16* krow = sK + (tn * 16 + l16) * 256 + h * 32 + lh * 16;
#pragma unroll
      for (int v = 0; v < 8; ++v)
        kf.u[v] = *(const unsigned*)(krow + 2 * v);
#pragma unroll
      for (int tm = 0; tm < 4; ++tm) {
        v8f z = {};
        sc[tm][tn] = WMMA_F16(qf[tm].h, kf.h, z);
      }
    }
    // dense relative-position bias (precomputed)
    const float* bh = biasd + h * 4096;
#pragma unroll
    for (int tm = 0; tm < 4; ++tm)
#pragma unroll
      for (int tn = 0; tn < 4; ++tn)
#pragma unroll
        for (int j = 0; j < 8; ++j) {
          const int p  = tm * 16 + j + lh * 8;
          const int qc = tn * 16 + l16;
          sc[tm][tn][j] += bh[p * 64 + qc];
        }
    // softmax over 64 columns per row
#pragma unroll
    for (int tm = 0; tm < 4; ++tm) {
#pragma unroll
      for (int j = 0; j < 8; ++j) {
        float r = sc[tm][0][j];
        r = fmaxf(r, sc[tm][1][j]);
        r = fmaxf(r, sc[tm][2][j]);
        r = fmaxf(r, sc[tm][3][j]);
        r = fmaxf(r, __shfl_xor(r, 1));
        r = fmaxf(r, __shfl_xor(r, 2));
        r = fmaxf(r, __shfl_xor(r, 4));
        r = fmaxf(r, __shfl_xor(r, 8));
        float ssum = 0.f;
#pragma unroll
        for (int tn = 0; tn < 4; ++tn) {
          float ex = __expf(sc[tm][tn][j] - r);
          sc[tm][tn][j] = ex;
          ssum += ex;
        }
        ssum += __shfl_xor(ssum, 1);
        ssum += __shfl_xor(ssum, 2);
        ssum += __shfl_xor(ssum, 4);
        ssum += __shfl_xor(ssum, 8);
        const float inv = 1.f / ssum;
#pragma unroll
        for (int tn = 0; tn < 4; ++tn) sc[tm][tn][j] *= inv;
      }
    }
    // out_h = attn @ v_h : stage each 16x64 band through LDS (C->A relayout)
    _Float16* as = sAS + wvs * 1024;
#pragma unroll
    for (int tm = 0; tm < 4; ++tm) {
#pragma unroll
      for (int tn = 0; tn < 4; ++tn)
#pragma unroll
        for (int j = 0; j < 8; ++j)
          as[(j + lh * 8) * 64 + tn * 16 + l16] = (_Float16)sc[tm][tn][j];

      v8f oacc[2];
      oacc[0] = {}; oacc[1] = {};
#pragma unroll
      for (int kc = 0; kc < 2; ++kc) {
        Frag af;
        const _Float16* arow = as + l16 * 64 + kc * 32;
#pragma unroll
        for (int v = 0; v < 8; ++v) {
          const int kk = lh * 16 + 2 * v;
          af.u[v] = *(const unsigned*)(arow + kk);
        }
#pragma unroll
        for (int tn2 = 0; tn2 < 2; ++tn2) {
          Frag bf;
          const _Float16* vrow =
              sVT + (h * 32 + tn2 * 16 + l16) * 64 + kc * 32 + lh * 16;
#pragma unroll
          for (int v = 0; v < 8; ++v)
            bf.u[v] = *(const unsigned*)(vrow + 2 * v);
          oacc[tn2] = WMMA_F16(af.h, bf.h, oacc[tn2]);
        }
      }
#pragma unroll
      for (int tn2 = 0; tn2 < 2; ++tn2)
#pragma unroll
        for (int j = 0; j < 8; ++j)
          sX[(tm * 16 + j + lh * 8) * 256 + h * 32 + tn2 * 16 + l16] =
              (_Float16)oacc[tn2][j];
    }
  }
  __syncthreads();

  // ---------------- Stage 4: proj = sO @ proj_w^T + proj_b ------------------
  // Wave-contiguous stream: frag index = wvs*16 + nti*8 + kt; depth-4 pipeline.
  float* og = out + (size_t)w * 16384;
  {
    Frag bbuf[4];
#pragma unroll
    for (int i = 0; i < 4; ++i) loadB(bbuf[i], wproj, wvs * 16 + i, lane);
    for (int nti = 0; nti < 2; ++nti) {
      const int nt = wvs * 2 + nti;
      const int e  = nt * 16 + l16;
      const _Float16* sxl = sX + opaque_zero_aligned();  // block A-frag hoisting
      v8f acc[4];
#pragma unroll
      for (int m = 0; m < 4; ++m) acc[m] = {};
#pragma unroll
      for (int kt = 0; kt < 8; ++kt) {
#pragma unroll
        for (int m = 0; m < 4; ++m) {
          Frag af;
          const _Float16* arow = sxl + (m * 16 + l16) * 256 + kt * 32;
#pragma unroll
          for (int v = 0; v < 8; ++v) {
            const int kk = lh * 16 + 2 * v;
            af.u[v] = *(const unsigned*)(arow + kk);
          }
          acc[m] = WMMA_F16(af.h, bbuf[kt & 3].h, acc[m]);
        }
        int s = nti * 8 + kt + 4;
        if (s > 15) s = 15;
        loadB(bbuf[kt & 3], wproj, wvs * 16 + s, lane);
      }
      const float pb = proj_b[e];
#pragma unroll
      for (int m = 0; m < 4; ++m)
#pragma unroll
        for (int j = 0; j < 8; ++j)
          og[(m * 16 + j + lh * 8) * 256 + e] = acc[m][j] + pb;
    }
  }
}

extern "C" void kernel_launch(void* const* d_in, const int* in_sizes, int n_in,
                              void* d_out, int out_size, void* d_ws, size_t ws_size,
                              hipStream_t stream) {
  (void)n_in; (void)ws_size; (void)out_size;
  const float* x       = (const float*)d_in[0];
  const float* noise   = (const float*)d_in[1];
  const float* gamma   = (const float*)d_in[2];
  const float* beta    = (const float*)d_in[3];
  const float* qkv_w   = (const float*)d_in[4];
  const float* proj_w  = (const float*)d_in[5];
  const float* proj_b  = (const float*)d_in[6];
  const float* biastab = (const float*)d_in[7];
  const int*   relidx  = (const int*)d_in[8];
  float* out = (float*)d_out;

  // Workspace layout: [0, 384K) qkv frags | [384K, 512K) proj frags |
  //                   [512K, 640K) dense bias
  unsigned* wsQ = (unsigned*)d_ws;                       // 48*8*32*8 dwords
  unsigned* wsP = (unsigned*)((char*)d_ws + 393216);     // 16*8*32*8 dwords
  float*    wsB = (float*)((char*)d_ws + 524288);        // 8*64*64 floats

  pack_w_kernel<<<48 * 8, 32, 0, stream>>>(qkv_w, wsQ, 48);
  pack_w_kernel<<<16 * 8, 32, 0, stream>>>(proj_w, wsP, 16);
  pack_bias_kernel<<<128, 256, 0, stream>>>(biastab, relidx, wsB);

  const int windows = in_sizes[0] / 16384;  // B*N = 2048
  lwa_kernel<<<windows, 256, 0, stream>>>(x, noise, gamma, beta,
                                          (const uint4*)wsQ, (const uint4*)wsP,
                                          proj_b, wsB, out);
}
